// ResidualQuantizer_29463475650671
// MI455X (gfx1250) — compile-verified, matrix operand order checked by reading, not run on hardware
//
#include <hip/hip_runtime.h>
#include <hip/hip_bf16.h>

typedef __attribute__((ext_vector_type(16))) _Float16 v16h;
typedef __attribute__((ext_vector_type(8)))  float    v8f;

#define DIM 32
#define KCB 256
#define NQ  3
#define TOK_TILE 16          // tokens per wave
#define WAVES_PER_BLOCK 8    // 256 threads
#define INV_TEMP 10.0f
#define GTAB 2048            // gumbel noise table entries (8 KB LDS)

union HV32 {
    uint4    u[2];
    v16h     v;
    _Float16 h[16];
};

__device__ __forceinline__ float hash01(unsigned int s) {
    s ^= s >> 17; s *= 0xed5ad4bbu;
    s ^= s >> 11; s *= 0xac4c1b51u;
    s ^= s >> 15; s *= 0x31848babu;
    s ^= s >> 14;
    return ((float)(s >> 8) + 0.5f) * (1.0f / 16777216.0f);  // (0,1)
}

__global__ void rq_init_loss_kernel(float* lossOut) {
    if (threadIdx.x < NQ) lossOut[threadIdx.x] = 0.0f;
}

__global__ __launch_bounds__(256)
void ResidualQuantizer_29463475650671_kernel(
    const float* __restrict__ x,
    const float* __restrict__ cb,
    float* __restrict__ outQ,
    float* __restrict__ outIdx,
    float* __restrict__ outLoss,
    int totalTokens, float invCount)
{
    __shared__ uint4 ldsB[NQ][KCB / 16][32][2];   // 48 KB: f16 B tiles
    __shared__ float ldsC2[NQ][KCB];              //  3 KB: entry norms
    __shared__ float ldsG[GTAB];                  //  8 KB: gumbel table
    __shared__ int   ldsIdx[WAVES_PER_BLOCK][TOK_TILE];

    const int tid  = threadIdx.x;
    const int wave = tid >> 5;
    const int lane = tid & 31;
    const int h    = lane >> 4;   // which K-set this lane owns
    const int m    = lane & 15;   // token row / codebook col within tile

    // ---- Phase 1a: gumbel table (once per block) ----
#pragma unroll
    for (int j = 0; j < GTAB / 256; ++j) {
        unsigned int i = (unsigned int)(tid + 256 * j);
        float u = hash01(i * 0x9E3779B9u + 0x7F4A7C15u);
        ldsG[tid + 256 * j] = -__logf(-__logf(u));
    }

    // ---- Phase 1b: pack codebooks into LDS (f16 B tiles + c2) ----
    {
        const float4* cb4 = (const float4*)cb;
#pragma unroll
        for (int q = 0; q < NQ; ++q) {
            const float4* e4 = cb4 + (size_t)(q * KCB + tid) * (DIM / 4);
            float d[DIM];
            float c2 = 0.0f;
#pragma unroll
            for (int j = 0; j < DIM / 4; ++j) {
                float4 v = e4[j];
                d[4 * j + 0] = v.x; d[4 * j + 1] = v.y;
                d[4 * j + 2] = v.z; d[4 * j + 3] = v.w;
            }
#pragma unroll
            for (int j = 0; j < DIM; ++j) c2 += d[j] * d[j];

            HV32 hv0, hv1;
#pragma unroll
            for (int j = 0; j < 8; ++j) {
                hv0.h[j]     = (_Float16)d[j];        // K = 0..7
                hv0.h[8 + j] = (_Float16)d[16 + j];   // K = 16..23
                hv1.h[j]     = (_Float16)d[8 + j];    // K = 8..15
                hv1.h[8 + j] = (_Float16)d[24 + j];   // K = 24..31
            }
            const int t = tid >> 4, s = tid & 15;
            ldsB[q][t][s][0]      = hv0.u[0];
            ldsB[q][t][s][1]      = hv0.u[1];
            ldsB[q][t][16 + s][0] = hv1.u[0];
            ldsB[q][t][16 + s][1] = hv1.u[1];
            ldsC2[q][tid] = c2;
        }
    }
    __syncthreads();

    // ---- Phase 2: per-wave token tile ----
    const int tile      = blockIdx.x * WAVES_PER_BLOCK + wave;
    const int tokenBase = tile * TOK_TILE;
    const int myToken   = tokenBase + m;

    float xorig[16], resid[16];
    {
        const float4* x4 = (const float4*)x + (size_t)myToken * (DIM / 4);
        float4 a0 = x4[h * 2 + 0];
        float4 a1 = x4[h * 2 + 1];
        float4 a2 = x4[4 + h * 2 + 0];
        float4 a3 = x4[4 + h * 2 + 1];
        xorig[0] = a0.x; xorig[1] = a0.y; xorig[2]  = a0.z; xorig[3]  = a0.w;
        xorig[4] = a1.x; xorig[5] = a1.y; xorig[6]  = a1.z; xorig[7]  = a1.w;
        xorig[8] = a2.x; xorig[9] = a2.y; xorig[10] = a2.z; xorig[11] = a2.w;
        xorig[12] = a3.x; xorig[13] = a3.y; xorig[14] = a3.z; xorig[15] = a3.w;
#pragma unroll
        for (int j = 0; j < 16; ++j) resid[j] = xorig[j];
    }

#pragma unroll
    for (int q = 0; q < NQ; ++q) {
        // A tile: 16x32 f16, one WMMA K-slab == full DIM
        v16h A;
#pragma unroll
        for (int j = 0; j < 16; ++j) A[j] = (_Float16)resid[j];

        // per-row noise seed bases: seed(tok,q,k) = (tok*3+q)*256 + k
        unsigned int baseSeed[8];
#pragma unroll
        for (int r = 0; r < 8; ++r) {
            unsigned int tok = (unsigned int)(tokenBase + h * 8 + r);
            baseSeed[r] = (tok * 3u + (unsigned int)q) * 256u + (unsigned int)m;
        }

        float best[8];
#pragma unroll
        for (int r = 0; r < 8; ++r) best[r] = -3.0e38f;

        // Sweep the 16 codebook column tiles: one WMMA each
#pragma unroll
        for (int t = 0; t < KCB / 16; ++t) {
            HV32 bv;
            bv.u[0] = ldsB[q][t][lane][0];
            bv.u[1] = ldsB[q][t][lane][1];
            v8f c = {};
            c = __builtin_amdgcn_wmma_f32_16x16x32_f16(
                    false, A, false, bv.v, (short)0, c, false, false);

            const unsigned int kcol = (unsigned int)(t * 16 + m); // C col N
            const float s2c = ldsC2[q][kcol] * (-INV_TEMP);       // -c2/T
            const unsigned int kofs = (unsigned int)(t * 16);
#pragma unroll
            for (int r = 0; r < 8; ++r) {
                unsigned int sd  = baseSeed[r] + kofs;
                unsigned int hsh = sd * 0x9E3779B1u;      // fibonacci hash
                float g = ldsG[hsh >> (32 - 11)];         // top 11 bits
                float score = fmaf(c[r], 2.0f * INV_TEMP, g + s2c);
                // pack candidate index into low mantissa bits -> 1-op argmax
                unsigned int p = (__float_as_uint(score) & 0xFFFFFF00u) | kcol;
                best[r] = fmaxf(best[r], __uint_as_float(p));
            }
        }

        // Argmax butterfly across the 16 lanes of each half
#pragma unroll
        for (int mask = 1; mask <= 8; mask <<= 1) {
#pragma unroll
            for (int r = 0; r < 8; ++r)
                best[r] = fmaxf(best[r], __shfl_xor(best[r], mask));
        }

        // Exchange indices so every lane sees the index of its own token
        if ((lane & 15) == 0) {
#pragma unroll
            for (int r = 0; r < 8; ++r)
                ldsIdx[wave][h * 8 + r] = (int)(__float_as_uint(best[r]) & 255u);
        }
        __syncthreads();
        const int ind = ldsIdx[wave][m];

        if (h == 0) outIdx[(size_t)myToken * NQ + q] = (float)ind;

        // Gather the selected codebook row (this lane's K-set) and update
        HV32 gv;
        gv.u[0] = ldsB[q][ind >> 4][h * 16 + (ind & 15)][0];
        gv.u[1] = ldsB[q][ind >> 4][h * 16 + (ind & 15)][1];
#pragma unroll
        for (int j = 0; j < 16; ++j) resid[j] -= (float)gv.h[j];

        // commit loss = mean(residual^2) after this stage
        float ss = 0.0f;
#pragma unroll
        for (int j = 0; j < 16; ++j) ss += resid[j] * resid[j];
#pragma unroll
        for (int mask = 1; mask <= 16; mask <<= 1) ss += __shfl_xor(ss, mask);
        if (lane == 0) atomicAdd(&outLoss[q], ss * invCount);
        __syncthreads();
    }

    // quantized_out = x - final residual
    {
        float4* o4 = (float4*)outQ + (size_t)myToken * (DIM / 4);
        float4 s0 = { xorig[0]  - resid[0],  xorig[1]  - resid[1],
                      xorig[2]  - resid[2],  xorig[3]  - resid[3] };
        float4 s1 = { xorig[4]  - resid[4],  xorig[5]  - resid[5],
                      xorig[6]  - resid[6],  xorig[7]  - resid[7] };
        float4 s2 = { xorig[8]  - resid[8],  xorig[9]  - resid[9],
                      xorig[10] - resid[10], xorig[11] - resid[11] };
        float4 s3 = { xorig[12] - resid[12], xorig[13] - resid[13],
                      xorig[14] - resid[14], xorig[15] - resid[15] };
        o4[h * 2 + 0]     = s0;
        o4[h * 2 + 1]     = s1;
        o4[4 + h * 2 + 0] = s2;
        o4[4 + h * 2 + 1] = s3;
    }
}

extern "C" void kernel_launch(void* const* d_in, const int* in_sizes, int n_in,
                              void* d_out, int out_size, void* d_ws, size_t ws_size,
                              hipStream_t stream) {
    const float* x  = (const float*)d_in[0];
    const float* cb = (const float*)d_in[1];

    const int totalTokens = in_sizes[0] / DIM;              // 8*32768
    float* outQ    = (float*)d_out;
    float* outIdx  = outQ + (size_t)totalTokens * DIM;
    float* outLoss = outIdx + (size_t)totalTokens * NQ;
    const float invCount = 1.0f / (float)((size_t)totalTokens * DIM);

    rq_init_loss_kernel<<<1, 32, 0, stream>>>(outLoss);

    const int tiles  = totalTokens / TOK_TILE;
    const int blocks = tiles / WAVES_PER_BLOCK;
    ResidualQuantizer_29463475650671_kernel<<<blocks, 32 * WAVES_PER_BLOCK, 0, stream>>>(
        x, cb, outQ, outIdx, outLoss, totalTokens, invCount);
}
